// InfoNCEWithQueue_70497593196766
// MI455X (gfx1250) — compile-verified, matrix operand order checked by reading, not run on hardware
//
#include <hip/hip_runtime.h>
#include <math.h>

// ---------------------------------------------------------------------------
// InfoNCE with queue: B=256, D=768, Q=65536.
// loss = -mean_i( sim[i,i] - logsumexp_j sim[i,j] ),  sim = (qn @ [kn;un]^T)/TEMP
//
//  K1: fused L2-normalize + fp32->bf16 (query -> qn, keys+queue -> Bm)
//  K2: bf16 WMMA GEMM (16x16x32), B tile staged through LDS with async
//      global->LDS (ASYNCcnt) double-buffered, fused row max / row sum-exp
//      partials per 128-column block + diagonal extraction
//  K3: per-row streaming log-sum-exp merge -> rowloss[i] = logZ_i - sim_ii
//  K4: mean over 256 rows -> d_out[0]
// ---------------------------------------------------------------------------

typedef __attribute__((ext_vector_type(16))) __bf16 v16bf;
typedef __attribute__((ext_vector_type(8)))  float  v8f;
typedef int vi4 __attribute__((vector_size(16)));          // matches builtin param

#define DIM      768
#define BROWS    256
#define QROWS    65536
#define NTOTAL   (BROWS + QROWS)   /* 65792 */
#define TILE_N   128
#define NBLK     (NTOTAL / TILE_N) /* 514 */
#define INV_TEMP 20.0f
#define LDSPITCH 40                /* ushorts per LDS B-row (32 + 8 pad) */

#if __has_builtin(__builtin_amdgcn_global_load_async_to_lds_b128) && \
    __has_builtin(__builtin_amdgcn_s_wait_asynccnt)
#define USE_ASYNC_LDS 1
#define GL_ASYNC_B128(gptr, lptr)                                   \
  __builtin_amdgcn_global_load_async_to_lds_b128(                   \
      (__attribute__((address_space(1))) vi4*)(gptr),               \
      (__attribute__((address_space(3))) vi4*)(lptr), 0, 0)
#else
#define USE_ASYNC_LDS 0
#endif

union Frag { uint4 q[2]; v16bf v; };

__device__ __forceinline__ unsigned short f2bf_rne(float f) {
  unsigned int u = __float_as_uint(f);
  unsigned int r = u + 0x7FFFu + ((u >> 16) & 1u);
  return (unsigned short)(r >> 16);
}

// ---------- K1: normalize + convert ---------------------------------------
__global__ void infonce_norm_bf16(const float* __restrict__ query,
                                  const float* __restrict__ keys,
                                  const float* __restrict__ queue,
                                  unsigned short* __restrict__ qn,
                                  unsigned short* __restrict__ Bm) {
  __shared__ float red[256];
  int b = blockIdx.x;
  const float* src;
  unsigned short* dst;
  if (b < 256)      { src = query + (size_t)b * DIM;          dst = qn + (size_t)b * DIM; }
  else if (b < 512) { src = keys  + (size_t)(b - 256) * DIM;  dst = Bm + (size_t)(b - 256) * DIM; }
  else              { src = queue + (size_t)(b - 512) * DIM;  dst = Bm + (size_t)(b - 512 + 256) * DIM; }

  float ss = 0.f;
  for (int k = threadIdx.x; k < DIM; k += 256) { float x = src[k]; ss += x * x; }
  red[threadIdx.x] = ss;
  __syncthreads();
  for (int s = 128; s > 0; s >>= 1) {
    if (threadIdx.x < s) red[threadIdx.x] += red[threadIdx.x + s];
    __syncthreads();
  }
  float scale = 1.0f / fmaxf(sqrtf(red[0]), 1e-8f);
  for (int k = threadIdx.x; k < DIM; k += 256)
    dst[k] = f2bf_rne(src[k] * scale);
}

// ---------- K2: WMMA GEMM (LDS-staged B) + softmax partials ---------------
__global__ __launch_bounds__(512)
void infonce_gemm_partials(const unsigned short* __restrict__ qn,
                           const unsigned short* __restrict__ Bm,
                           float* __restrict__ pM,
                           float* __restrict__ pS,
                           float* __restrict__ diag) {
  __shared__ __align__(16) unsigned short ldsB[2][TILE_N * LDSPITCH];

  const int t    = threadIdx.x;
  const int wave = t >> 5;                 // 0..15 -> M tile
  const int lane = t & 31;
  const int l16  = lane & 15;
  const int half = lane >> 4;              // 0: K 0..7/16..23, 1: K 8..15/24..31
  const int mrow = wave * 16 + l16;        // A-fragment row for this lane
  const int nbase = blockIdx.x * TILE_N;

  // Cooperative B-tile copy: thread t moves 16B segment (row t/4, seg t%4)
  const int crow = t >> 2;                 // 0..127
  const int cseg = t & 3;                  // 0..3 (8 bf16 each)
  const unsigned short* gsrc = Bm + (size_t)(nbase + crow) * DIM + cseg * 8;
  unsigned short* ldst0 = &ldsB[0][crow * LDSPITCH + cseg * 8];
  unsigned short* ldst1 = &ldsB[1][crow * LDSPITCH + cseg * 8];

  const unsigned short* aptr = qn + (size_t)mrow * DIM;

  v8f acc[8];
#pragma unroll
  for (int j = 0; j < 8; ++j) acc[j] = (v8f){0.f,0.f,0.f,0.f,0.f,0.f,0.f,0.f};

  // Prime buffer 0 with K-chunk 0.
#if USE_ASYNC_LDS
  GL_ASYNC_B128(gsrc, ldst0);
  __builtin_amdgcn_s_wait_asynccnt(0);
#else
  *(uint4*)ldst0 = *(const uint4*)gsrc;
#endif
  __syncthreads();

  int p = 0;
  for (int kk = 0; kk < DIM; kk += 32) {
    // Prefetch next K-chunk into the other buffer (overlaps with WMMAs below).
    if (kk + 32 < DIM) {
      const unsigned short* gnext = gsrc + (kk + 32);
      unsigned short* lnext = p ? ldst0 : ldst1;
#if USE_ASYNC_LDS
      GL_ASYNC_B128(gnext, lnext);
#else
      *(uint4*)lnext = *(const uint4*)gnext;
#endif
    }

    Frag a;
    a.q[0] = *(const uint4*)(aptr + kk + half * 8);
    a.q[1] = *(const uint4*)(aptr + kk + 16 + half * 8);
#pragma unroll
    for (int j = 0; j < 8; ++j) {
      const unsigned short* lrow = &ldsB[p][(j * 16 + l16) * LDSPITCH];
      Frag b;
      b.q[0] = *(const uint4*)(lrow + half * 8);
      b.q[1] = *(const uint4*)(lrow + 16 + half * 8);
      acc[j] = __builtin_amdgcn_wmma_f32_16x16x32_bf16(
          false, a.v, false, b.v, (short)0, acc[j], false, false);
    }

#if USE_ASYNC_LDS
    __builtin_amdgcn_s_wait_asynccnt(0);
#endif
    __syncthreads();
    p ^= 1;
  }

  // C layout: lane(0-15): M = r, col = l16 ; lane(16-31): M = r+8, col = l16.
#pragma unroll
  for (int r = 0; r < 8; ++r) {
    const int grow = wave * 16 + r + 8 * half;   // global row of acc[*][r]

    float mx = -INFINITY;
#pragma unroll
    for (int j = 0; j < 8; ++j) mx = fmaxf(mx, acc[j][r] * INV_TEMP);
    for (int msk = 1; msk < 16; msk <<= 1)
      mx = fmaxf(mx, __shfl_xor(mx, msk, 32));

    float sum = 0.f;
#pragma unroll
    for (int j = 0; j < 8; ++j) sum += __expf(acc[j][r] * INV_TEMP - mx);
    for (int msk = 1; msk < 16; msk <<= 1)
      sum += __shfl_xor(sum, msk, 32);

    if (l16 == 0) {
      pM[(size_t)grow * NBLK + blockIdx.x] = mx;
      pS[(size_t)grow * NBLK + blockIdx.x] = sum;
    }

    // Diagonal sim[i,i] lives in the first 256 columns (keys block).
#pragma unroll
    for (int j = 0; j < 8; ++j) {
      int gcol = nbase + j * 16 + l16;
      if (gcol == grow) diag[grow] = acc[j][r] * INV_TEMP;
    }
  }
}

// ---------- K3: per-row log-sum-exp merge ---------------------------------
__global__ void infonce_rowreduce(const float* __restrict__ pM,
                                  const float* __restrict__ pS,
                                  const float* __restrict__ diag,
                                  float* __restrict__ rowloss) {
  __shared__ float red[256];
  const int row = blockIdx.x;
  const float* pm = pM + (size_t)row * NBLK;
  const float* ps = pS + (size_t)row * NBLK;

  float mx = -INFINITY;
  for (int i = threadIdx.x; i < NBLK; i += 256) mx = fmaxf(mx, pm[i]);
  red[threadIdx.x] = mx;
  __syncthreads();
  for (int s = 128; s > 0; s >>= 1) {
    if (threadIdx.x < s) red[threadIdx.x] = fmaxf(red[threadIdx.x], red[threadIdx.x + s]);
    __syncthreads();
  }
  const float M = red[0];
  __syncthreads();

  float sum = 0.f;
  for (int i = threadIdx.x; i < NBLK; i += 256) sum += ps[i] * __expf(pm[i] - M);
  red[threadIdx.x] = sum;
  __syncthreads();
  for (int s = 128; s > 0; s >>= 1) {
    if (threadIdx.x < s) red[threadIdx.x] += red[threadIdx.x + s];
    __syncthreads();
  }
  if (threadIdx.x == 0)
    rowloss[row] = (M + logf(red[0])) - diag[row];
}

// ---------- K4: mean ------------------------------------------------------
__global__ void infonce_finalize(const float* __restrict__ rowloss,
                                 float* __restrict__ out) {
  __shared__ float red[256];
  red[threadIdx.x] = rowloss[threadIdx.x];
  __syncthreads();
  for (int s = 128; s > 0; s >>= 1) {
    if (threadIdx.x < s) red[threadIdx.x] += red[threadIdx.x + s];
    __syncthreads();
  }
  if (threadIdx.x == 0) out[0] = red[0] / 256.0f;
}

// ---------------------------------------------------------------------------
extern "C" void kernel_launch(void* const* d_in, const int* in_sizes, int n_in,
                              void* d_out, int out_size, void* d_ws, size_t ws_size,
                              hipStream_t stream) {
  (void)in_sizes; (void)n_in; (void)out_size; (void)ws_size;

  const float* query = (const float*)d_in[0];
  const float* keys  = (const float*)d_in[1];
  const float* queue = (const float*)d_in[2];
  float* out = (float*)d_out;

  // Workspace layout (all chunks 16B-aligned by construction)
  char* ws = (char*)d_ws;
  size_t offB    = 0;                                             // bf16 [65792 x 768]
  size_t offQ    = offB  + (size_t)NTOTAL * DIM * sizeof(unsigned short);
  size_t offPM   = offQ  + (size_t)BROWS  * DIM * sizeof(unsigned short);
  size_t offPS   = offPM + (size_t)BROWS * NBLK * sizeof(float);
  size_t offDiag = offPS + (size_t)BROWS * NBLK * sizeof(float);
  size_t offRL   = offDiag + (size_t)BROWS * sizeof(float);

  unsigned short* Bm = (unsigned short*)(ws + offB);
  unsigned short* qn = (unsigned short*)(ws + offQ);
  float* pM   = (float*)(ws + offPM);
  float* pS   = (float*)(ws + offPS);
  float* diag = (float*)(ws + offDiag);
  float* rl   = (float*)(ws + offRL);

  infonce_norm_bf16<<<512 + QROWS, 256, 0, stream>>>(query, keys, queue, qn, Bm);
  infonce_gemm_partials<<<NBLK, 512, 0, stream>>>(qn, Bm, pM, pS, diag);
  infonce_rowreduce<<<BROWS, 256, 0, stream>>>(pM, pS, diag, rl);
  infonce_finalize<<<1, 256, 0, stream>>>(rl, out);
}
